// LiDAREncoder_67963562492441
// MI455X (gfx1250) — matrix operand order repaired and checked by reading, not compiled, not run on hardware
//
#include <hip/hip_runtime.h>
#include <hip/hip_bf16.h>

typedef __attribute__((ext_vector_type(16))) _Float16 v16h;
typedef __attribute__((ext_vector_type(8)))  float    v8f;

#define NPTS    200000
#define NTILES  (NPTS / 16)      // 12500 exact
#define HWGRID  62500            // 250*250
#define BSTRIDE 16000000         // 256*62500
#define OUT_N   64000000LL       // 4*256*250*250

// ---------------- zero fill (float4 grid-stride) ----------------
__global__ void __launch_bounds__(256) k_zero(float* __restrict__ p, long long n) {
  long long i = (long long)blockIdx.x * blockDim.x + threadIdx.x;
  long long stride = (long long)gridDim.x * blockDim.x;
  float4* p4 = (float4*)p;
  long long n4 = n >> 2;
  for (long long j = i; j < n4; j += stride) p4[j] = make_float4(0.f, 0.f, 0.f, 0.f);
}

// ---------------- one-time weight prep: fp32 -> f16, pre-swizzled WMMA B layout ----------
// WB[((nt*nslices + s)*32 + lane)*16 + e] = (f16) W[n*K + k]
//   n = nt*16 + (lane&15),  k = 32*s + 8*(lane>>4) + (e&7) + 16*(e>>3)
__global__ void __launch_bounds__(256) k_prep_w(const float* __restrict__ W,
                                               _Float16* __restrict__ WB,
                                               int nslices, int Kdim, int total)
{
  int idx = blockIdx.x * 256 + threadIdx.x;
  if (idx >= total) return;
  int e    = idx & 15;
  int lane = (idx >> 4) & 31;
  int g    = idx >> 9;            // nt*nslices + s
  int s    = g % nslices;
  int nt   = g / nslices;
  int n = nt * 16 + (lane & 15);
  int k = 32 * s + 8 * (lane >> 4) + (e & 7) + 16 * (e >> 3);
  WB[idx] = (_Float16)W[(size_t)n * Kdim + k];
}

// ---------------- layer 1: 7 -> 64 scalar MLP + voxel meta + BN stats ----------------
__global__ void __launch_bounds__(256) k_layer1(
    const float* __restrict__ pts, const int* __restrict__ bidx,
    const float* __restrict__ W1, const float* __restrict__ b1,
    _Float16* __restrict__ X1, int* __restrict__ meta,
    float* __restrict__ sum1, float* __restrict__ sumsq1)
{
  __shared__ float lsum[64];
  __shared__ float lssq[64];
  const int t = threadIdx.x;
  if (t < 64) { lsum[t] = 0.f; lssq[t] = 0.f; }
  __syncthreads();

  const int p = blockIdx.x * 256 + t;
  if (p < NPTS) {
    const float* q = pts + (size_t)p * 7;
    float in[7];
    float x = (q[0] - (-50.0f)) / 0.4f;
    float y = (q[1] - (-50.0f)) / 0.4f;
    float z = (q[2] - (-3.0f))  / 0.4f;
    in[0] = x; in[1] = y; in[2] = z;
    in[3] = q[3]; in[4] = q[4]; in[5] = q[5]; in[6] = q[6];
    int gx = min(max((int)x, 0), 249);
    int gy = min(max((int)y, 0), 249);
    int b  = bidx[p];
    meta[p] = b * BSTRIDE + gx * 250 + gy;
    #pragma unroll 4
    for (int j = 0; j < 64; ++j) {
      float acc = b1[j];
      #pragma unroll
      for (int k = 0; k < 7; ++k) acc = fmaf(in[k], W1[j * 7 + k], acc);
      X1[(size_t)p * 64 + j] = (_Float16)acc;
      atomicAdd(&lsum[j], acc);
      atomicAdd(&lssq[j], acc * acc);
    }
  }
  __syncthreads();
  if (t < 64) {
    atomicAdd(&sum1[t],   lsum[t]);
    atomicAdd(&sumsq1[t], lssq[t]);
  }
}

// ---------------- fold BN stats into per-channel scale/shift ----------------
__global__ void k_finalize(const float* __restrict__ sum, const float* __restrict__ sumsq,
                           const float* __restrict__ g, const float* __restrict__ be,
                           float* __restrict__ scale, float* __restrict__ shift, int nch)
{
  int j = blockIdx.x * blockDim.x + threadIdx.x;
  if (j < nch) {
    const float inv_n = 1.0f / (float)NPTS;
    float mean = sum[j] * inv_n;
    float var  = sumsq[j] * inv_n - mean * mean;
    float sc   = g[j] * rsqrtf(var + 1e-5f);
    scale[j] = sc;
    shift[j] = be[j] - mean * sc;
  }
}

// ---------------- layer 2: BN1+ReLU then [200000x64] @ [64x128] via WMMA + BN2 stats ----------------
__global__ void __launch_bounds__(256) k_layer2(
    const _Float16* __restrict__ X1, const _Float16* __restrict__ WB2,
    const float* __restrict__ b2,
    const float* __restrict__ scale1, const float* __restrict__ shift1,
    _Float16* __restrict__ X2, float* __restrict__ sum2, float* __restrict__ sumsq2)
{
  const int lane = threadIdx.x & 31;
  const int wv   = threadIdx.x >> 5;
  const int tile = blockIdx.x * 8 + wv;
  if (tile >= NTILES) return;          // wave-uniform: WMMA waves keep EXEC all-ones
  const int hi = lane >> 4;            // half-wave select
  const int m  = lane & 15;
  const int pt = tile * 16 + m;        // A-matrix row owned by this lane
  const v16h* Bv = (const v16h*)WB2;   // pre-swizzled f16 B tiles

  // prefetch next M-tile's activation row (global_prefetch_b8)
  if (tile + 1 < NTILES) __builtin_prefetch(X1 + (size_t)(pt + 16) * 64, 0, 1);

  // A tile: 16x64 (two 16x32 K-slices), ISA 16-bit A layout, BN1+ReLU applied inline
  v16h a[2];
  #pragma unroll
  for (int s = 0; s < 2; ++s) {
    const _Float16* src = X1 + (size_t)pt * 64 + 32 * s + 8 * hi;
    #pragma unroll
    for (int j = 0; j < 8; ++j) {
      const int k0 = 32 * s + 8 * hi + j;          // elements 0..7
      const int k1 = k0 + 16;                      // elements 8..15
      float h0 = fmaxf(fmaf((float)src[j],      scale1[k0], shift1[k0]), 0.f);
      float h1 = fmaxf(fmaf((float)src[16 + j], scale1[k1], shift1[k1]), 0.f);
      a[s][j]     = (_Float16)h0;
      a[s][j + 8] = (_Float16)h1;
    }
  }

  for (int nt = 0; nt < 8; ++nt) {
    const int n = nt * 16 + m;                     // output channel owned by this lane
    v8f acc;
    const float bias = b2[n];
    #pragma unroll
    for (int r = 0; r < 8; ++r) acc[r] = bias;
    #pragma unroll
    for (int s = 0; s < 2; ++s) {
      v16h bm = Bv[(nt * 2 + s) * 32 + lane];      // one aligned 32B load per lane
      acc = __builtin_amdgcn_wmma_f32_16x16x32_f16(false, a[s], false, bm,
                                                   (short)0, acc, false, false);
    }
    float ls = 0.f, lss = 0.f;
    #pragma unroll
    for (int r = 0; r < 8; ++r) {                  // D: point = tile*16 + r + 8h
      const int ptd = tile * 16 + r + 8 * hi;
      float v = acc[r];
      X2[(size_t)ptd * 128 + n] = (_Float16)v;
      ls += v; lss += v * v;
    }
    // lanes L and L+16 own the same channel n: combine across half-waves first
    ls  += __shfl_xor(ls, 16, 32);
    lss += __shfl_xor(lss, 16, 32);
    if (hi == 0) {
      atomicAdd(&sum2[n], ls);
      atomicAdd(&sumsq2[n], lss);
    }
  }
}

// ---------------- layer 3: BN2+ReLU then [200000x128] @ [128x256] via WMMA + scatter-max ----------------
__global__ void __launch_bounds__(256) k_layer3(
    const _Float16* __restrict__ X2, const _Float16* __restrict__ WB3,
    const float* __restrict__ b3,
    const float* __restrict__ scale2, const float* __restrict__ shift2,
    const int* __restrict__ meta, float* __restrict__ bev)
{
  const int lane = threadIdx.x & 31;
  const int wv   = threadIdx.x >> 5;
  const int tile = blockIdx.x * 8 + wv;
  if (tile >= NTILES) return;
  const int hi = lane >> 4;
  const int m  = lane & 15;
  const int pt = tile * 16 + m;
  const v16h* Bv = (const v16h*)WB3;

  if (tile + 1 < NTILES) __builtin_prefetch(X2 + (size_t)(pt + 16) * 128, 0, 1);

  // A tile: 16x128 (four 16x32 K-slices), BN2+ReLU inline
  v16h a[4];
  #pragma unroll
  for (int s = 0; s < 4; ++s) {
    const _Float16* src = X2 + (size_t)pt * 128 + 32 * s + 8 * hi;
    #pragma unroll
    for (int j = 0; j < 8; ++j) {
      const int k0 = 32 * s + 8 * hi + j;
      const int k1 = k0 + 16;
      float h0 = fmaxf(fmaf((float)src[j],      scale2[k0], shift2[k0]), 0.f);
      float h1 = fmaxf(fmaf((float)src[16 + j], scale2[k1], shift2[k1]), 0.f);
      a[s][j]     = (_Float16)h0;
      a[s][j + 8] = (_Float16)h1;
    }
  }

  // destination voxel bases for the 8 points this lane holds in D
  int base[8];
  #pragma unroll
  for (int r = 0; r < 8; ++r) base[r] = meta[tile * 16 + r + 8 * hi];

  for (int nt = 0; nt < 16; ++nt) {
    const int n = nt * 16 + m;
    v8f acc;
    const float bias = b3[n];
    #pragma unroll
    for (int r = 0; r < 8; ++r) acc[r] = bias;
    #pragma unroll
    for (int s = 0; s < 4; ++s) {
      v16h bm = Bv[(nt * 4 + s) * 32 + lane];
      acc = __builtin_amdgcn_wmma_f32_16x16x32_f16(false, a[s], false, bm,
                                                   (short)0, acc, false, false);
    }
    // scatter-max: result is max(segment_max, 0) with zero-init grid, so only
    // positive features matter; non-negative IEEE floats order correctly as uints.
    const size_t cofs = (size_t)n * HWGRID;
    #pragma unroll
    for (int r = 0; r < 8; ++r) {
      float v = acc[r];
      if (v > 0.f) {
        atomicMax((unsigned int*)(bev + (size_t)base[r] + cofs), __float_as_uint(v));
      }
    }
  }
}

extern "C" void kernel_launch(void* const* d_in, const int* in_sizes, int n_in,
                              void* d_out, int out_size, void* d_ws, size_t ws_size,
                              hipStream_t stream) {
  const float* points = (const float*)d_in[0];
  const int*   bidx   = (const int*)  d_in[1];
  const float* W1 = (const float*)d_in[2];
  const float* b1 = (const float*)d_in[3];
  const float* g1 = (const float*)d_in[4];
  const float* be1= (const float*)d_in[5];
  const float* W2 = (const float*)d_in[6];
  const float* b2 = (const float*)d_in[7];
  const float* g2 = (const float*)d_in[8];
  const float* be2= (const float*)d_in[9];
  const float* W3 = (const float*)d_in[10];
  const float* b3 = (const float*)d_in[11];
  float* bev = (float*)d_out;

  // workspace carve-up (all slab sizes keep >=32B alignment)
  char* ws = (char*)d_ws;
  _Float16* X1  = (_Float16*)ws; ws += (size_t)NPTS * 64  * sizeof(_Float16);  // 25.6 MB
  _Float16* X2  = (_Float16*)ws; ws += (size_t)NPTS * 128 * sizeof(_Float16);  // 51.2 MB
  int*      meta= (int*)ws;      ws += (size_t)NPTS * sizeof(int);             // 0.8 MB
  _Float16* WB2 = (_Float16*)ws; ws += (size_t)8  * 2 * 32 * 16 * sizeof(_Float16); // 16 KB
  _Float16* WB3 = (_Float16*)ws; ws += (size_t)16 * 4 * 32 * 16 * sizeof(_Float16); // 64 KB
  float* stats = (float*)ws;                                                   // 768 floats
  float* sum1   = stats +   0; float* sumsq1 = stats +  64;
  float* scale1 = stats + 128; float* shift1 = stats + 192;
  float* sum2   = stats + 256; float* sumsq2 = stats + 384;
  float* scale2 = stats + 512; float* shift2 = stats + 640;

  const int TOT2 = 8 * 2 * 32 * 16;    // 8192
  const int TOT3 = 16 * 4 * 32 * 16;   // 32768

  // 1) zero accumulators and output grid (atomics accumulate into both)
  k_zero<<<1, 256, 0, stream>>>(stats, 768);
  k_zero<<<4096, 256, 0, stream>>>(bev, OUT_N);

  // 2) pre-swizzle weights into f16 WMMA B layout (once per launch)
  k_prep_w<<<(TOT2 + 255) / 256, 256, 0, stream>>>(W2, WB2, 2, 64,  TOT2);
  k_prep_w<<<(TOT3 + 255) / 256, 256, 0, stream>>>(W3, WB3, 4, 128, TOT3);

  // 3) per-point 7->64 MLP, voxel meta, BN1 raw stats
  k_layer1<<<(NPTS + 255) / 256, 256, 0, stream>>>(points, bidx, W1, b1, X1, meta, sum1, sumsq1);
  k_finalize<<<1, 64, 0, stream>>>(sum1, sumsq1, g1, be1, scale1, shift1, 64);

  // 4) 64->128 WMMA GEMM with fused BN1+ReLU, BN2 raw stats
  k_layer2<<<(NTILES + 7) / 8, 256, 0, stream>>>(X1, WB2, b2, scale1, shift1, X2, sum2, sumsq2);
  k_finalize<<<1, 128, 0, stream>>>(sum2, sumsq2, g2, be2, scale2, shift2, 128);

  // 5) 128->256 WMMA GEMM with fused BN2+ReLU, scatter-max into BEV
  k_layer3<<<(NTILES + 7) / 8, 256, 0, stream>>>(X2, WB3, b3, scale2, shift2, meta, bev);
}
